// Gemma2Attention_40063454937445
// MI455X (gfx1250) — matrix operand order repaired and checked by reading, not compiled, not run on hardware
//
#include <hip/hip_runtime.h>
#include <hip/hip_bf16.h>

// ---------------------------------------------------------------------------
// Gemma2 attention layer for MI455X (gfx1250, wave32, WMMA + TDM).
// Pipeline: fp32->bf16 convert, QKV GEMM (bf16 WMMA, TDM-staged A tiles),
// RoPE, sliding-window flash attention (bf16 WMMA, fp32 softmax, TDM-staged
// K tiles), output projection GEMM (fp32 out).
// ---------------------------------------------------------------------------

typedef __attribute__((ext_vector_type(16))) __bf16 v16bf;
typedef __attribute__((ext_vector_type(2)))  __bf16 bf2;
typedef __attribute__((ext_vector_type(8)))  float  v8f;
typedef __attribute__((ext_vector_type(4)))  unsigned int u32x4;
typedef __attribute__((ext_vector_type(4)))  int i32x4;
typedef __attribute__((ext_vector_type(8)))  int i32x8;

#define S_LEN 4096
#define EMB   3584
#define HQ    16
#define HKV   8
#define HD    256
#define WIN   1024

#if defined(__has_builtin)
#if __has_builtin(__builtin_amdgcn_tensor_load_to_lds) && \
    __has_builtin(__builtin_amdgcn_s_wait_tensorcnt)
#define HAVE_TDM 1
#endif
#endif
#ifndef HAVE_TDM
#define HAVE_TDM 0
#endif

__device__ __forceinline__ v8f wmma_bf16(v16bf a, v16bf b, v8f c) {
    // D = A(16x32 bf16) * B(32x16 bf16) + C(16x16 f32)
    return __builtin_amdgcn_wmma_f32_16x16x32_bf16(
        /*neg_a=*/false, a, /*neg_b=*/false, b,
        /*c_mod=*/(short)0, c, /*reuse_a=*/false, /*reuse_b=*/false);
}

// A-matrix (16x32, 16-bit) K-offset for VGPR pair vg (0..7), lane-half hf:
// lanes 0-15: K = {0..7, 16..23}; lanes 16-31: +8  (ISA 7.12.2)
__device__ __forceinline__ int a_koff(int vg, int hf) {
    return (vg < 4 ? 2 * vg : 16 + 2 * (vg - 4)) + hf * 8;
}
// B-matrix (32x16, 16-bit): same K order as A: lanes 0-15 K = {0..7, 16..23}; lanes 16-31: +8
__device__ __forceinline__ int b_koff(int vg, int hf) {
    return (vg < 4 ? 2 * vg : 16 + 2 * (vg - 4)) + hf * 8;
}

// ---------------------------------------------------------------------------
// Tensor Data Mover: 2-D tile (16-bit elements) global -> LDS, one descriptor.
// D# group 0: count | lds_addr | global_addr | type=2 (ISA 8.3)
// D# group 1: data_size=2B, tensor_dim0/1, tile_dim0/1, dim0 stride (ISA 8.4)
// ---------------------------------------------------------------------------
__device__ __forceinline__ void tdm_load_2d_bf16(
    unsigned lds_off, const void* gaddr,
    unsigned tile_d0, unsigned tile_d1,
    unsigned tensor_d0, unsigned tensor_d1,
    unsigned long long row_stride_elems) {
#if HAVE_TDM
    unsigned long long g = (unsigned long long)(size_t)gaddr;
    u32x4 g0;
    g0.x = 1u;                                    // count=1 (valid user D#)
    g0.y = lds_off;                               // lds_addr (bytes)
    g0.z = (unsigned)g;                           // global_addr[31:0]
    g0.w = (unsigned)((g >> 32) & 0x01FFFFFFu) |  // global_addr[56:32]
           (2u << 30);                            // type=2 ("image")
    i32x8 g1;
    g1[0] = (int)(1u << 16);                      // data_size=1 -> 2 bytes
    g1[1] = (int)((tensor_d0 & 0xFFFFu) << 16);   // tensor_dim0 @ bit48
    g1[2] = (int)(((tensor_d0 >> 16) & 0xFFFFu) | ((tensor_d1 & 0xFFFFu) << 16));
    g1[3] = (int)(((tensor_d1 >> 16) & 0xFFFFu) | ((tile_d0 & 0xFFFFu) << 16));
    g1[4] = (int)(tile_d1 & 0xFFFFu);             // tile_dim1 | tile_dim2=0
    g1[5] = (int)(unsigned)(row_stride_elems & 0xFFFFFFFFull);
    g1[6] = (int)(unsigned)((row_stride_elems >> 32) & 0xFFFFull); // stride1=0
    g1[7] = 0;
    i32x4 z4 = {0, 0, 0, 0};
#if __clang_major__ >= 23
    i32x8 z8 = {0, 0, 0, 0, 0, 0, 0, 0};
    __builtin_amdgcn_tensor_load_to_lds(g0, g1, z4, z4, z8, 0);
#else
    __builtin_amdgcn_tensor_load_to_lds(g0, g1, z4, z4, 0);
#endif
#else
    (void)lds_off; (void)gaddr; (void)tile_d0; (void)tile_d1;
    (void)tensor_d0; (void)tensor_d1; (void)row_stride_elems;
#endif
}

__device__ __forceinline__ void tdm_wait() {
#if HAVE_TDM
    __builtin_amdgcn_s_wait_tensorcnt(0);
#endif
}

// ---------------------------------------------------------------------------
// fp32 -> bf16 conversion
// ---------------------------------------------------------------------------
__global__ void cvt_f32_bf16(const float* __restrict__ in,
                             __bf16* __restrict__ out, int n) {
    int i = (blockIdx.x * blockDim.x + threadIdx.x) * 4;
    if (i + 3 < n) {
        float4 v = *(const float4*)(in + i);
        out[i + 0] = (__bf16)v.x;
        out[i + 1] = (__bf16)v.y;
        out[i + 2] = (__bf16)v.z;
        out[i + 3] = (__bf16)v.w;
    } else {
        for (; i < n; ++i) out[i] = (__bf16)in[i];
    }
}

// ---------------------------------------------------------------------------
// RoPE, in place on bf16 [S, nh*256]; each thread handles the (i, i+128) pair.
// ---------------------------------------------------------------------------
__global__ void rope_kernel(__bf16* __restrict__ X,
                            const int* __restrict__ pos_ids, int nheads) {
    int tid = blockIdx.x * blockDim.x + threadIdx.x;
    int total = S_LEN * nheads * 128;
    if (tid >= total) return;
    int i   = tid & 127;
    int hh  = (tid >> 7) % nheads;
    int row = tid / (128 * nheads);
    float pos = (float)pos_ids[row];
    float inv = __powf(10000.0f, -(float)i * (1.0f / 128.0f));
    float fr  = pos * inv;
    float sv, cv;
    __sincosf(fr, &sv, &cv);
    __bf16* p = X + (size_t)row * nheads * HD + hh * HD;
    float x1 = (float)p[i];
    float x2 = (float)p[i + 128];
    p[i]       = (__bf16)(x1 * cv - x2 * sv);
    p[i + 128] = (__bf16)(x2 * cv + x1 * sv);
}

// ---------------------------------------------------------------------------
// Tiled bf16 WMMA GEMM: C[M,N] = A[M,K] * B[K,N]
// Block tile 128x64, 8 waves. A tile staged by the Tensor Data Mover (wave 0
// issues the descriptor; TENSORcnt + barrier orders it), B staged transposed
// by all threads so every B-fragment load is one b32 LDS access.
// ---------------------------------------------------------------------------
template <bool OUT_BF16>
__global__ __launch_bounds__(256) void gemm_wmma_bf16(
    const __bf16* __restrict__ A, const __bf16* __restrict__ B,
    void* __restrict__ Cout, int M, int N, int Kd) {
    __shared__ __bf16 Alds[128 * 32];   // [row][k], stride 32 (TDM layout)
    __shared__ __bf16 Blds[64 * 32];    // transposed: [n][k], stride 32

    const int wave = threadIdx.x >> 5;
    const int lh   = threadIdx.x & 15;
    const int hf   = (threadIdx.x >> 4) & 1;
    const int mBase = blockIdx.y * 128;
    const int nBase = blockIdx.x * 64;
    const unsigned alds_off = (unsigned)(size_t)(void*)Alds;

    v8f acc[4] = {{0,0,0,0,0,0,0,0}, {0,0,0,0,0,0,0,0},
                  {0,0,0,0,0,0,0,0}, {0,0,0,0,0,0,0,0}};

    for (int kb = 0; kb < Kd; kb += 32) {
        __syncthreads();   // previous tile fully consumed
#if HAVE_TDM
        if (wave == 0) {   // async DMA: A tile 128x32, row stride Kd elems
            tdm_load_2d_bf16(alds_off, A + (size_t)mBase * Kd + kb,
                             /*tile*/ 32, 128, /*tensor*/ 32, 128,
                             (unsigned long long)Kd);
        }
#else
#pragma unroll
        for (int it = 0; it < 2; ++it) {
            int idx = threadIdx.x + it * 256;      // 512 chunks of 8 bf16
            int row = idx >> 2;
            int kc  = (idx & 3) * 8;
            *(uint4*)(Alds + row * 32 + kc) =
                *(const uint4*)(A + (size_t)(mBase + row) * Kd + kb + kc);
        }
#endif
        if (kb + 32 < Kd)  // global_prefetch_b8 hint for the next B tile
            __builtin_prefetch(B + (size_t)(kb + 32 + (threadIdx.x & 31)) * N + nBase, 0, 1);
        // ---- stage B transposed (64 n x 32 k), overlapped with the TDM ----
#pragma unroll
        for (int it = 0; it < 2; ++it) {
            int idx  = threadIdx.x + it * 256;     // 512 chunks of 4 bf16 (N)
            int krow = idx >> 4;
            int nc   = (idx & 15) * 4;
            const __bf16* gsrc = B + (size_t)(kb + krow) * N + nBase + nc;
            bf2 a0 = *(const bf2*)(gsrc);
            bf2 a1 = *(const bf2*)(gsrc + 2);
            Blds[(nc + 0) * 32 + krow] = a0.x;
            Blds[(nc + 1) * 32 + krow] = a0.y;
            Blds[(nc + 2) * 32 + krow] = a1.x;
            Blds[(nc + 3) * 32 + krow] = a1.y;
        }
#if HAVE_TDM
        if (wave == 0) tdm_wait();
#endif
        __syncthreads();

        // ---- A fragment (16x32) ----
        v16bf af;
        int arow = wave * 16 + lh;
#pragma unroll
        for (int vg = 0; vg < 8; ++vg) {
            bf2 p = *(const bf2*)(Alds + arow * 32 + a_koff(vg, hf));
            af[2 * vg]     = p.x;
            af[2 * vg + 1] = p.y;
        }
        // ---- 4 WMMAs over the 64-wide N tile ----
#pragma unroll
        for (int j = 0; j < 4; ++j) {
            v16bf bfr;
#pragma unroll
            for (int vg = 0; vg < 8; ++vg) {
                bf2 p = *(const bf2*)(Blds + (j * 16 + lh) * 32 + b_koff(vg, hf));
                bfr[2 * vg]     = p.x;
                bfr[2 * vg + 1] = p.y;
            }
            acc[j] = wmma_bf16(af, bfr, acc[j]);
        }
    }

    // ---- epilogue: C layout — VGPR e: lanes0-15 M=e, lanes16-31 M=8+e ----
#pragma unroll
    for (int j = 0; j < 4; ++j) {
#pragma unroll
        for (int e = 0; e < 8; ++e) {
            int row = mBase + wave * 16 + (hf ? 8 + e : e);
            int col = nBase + j * 16 + lh;
            if constexpr (OUT_BF16)
                ((__bf16*)Cout)[(size_t)row * N + col] = (__bf16)acc[j][e];
            else
                ((float*)Cout)[(size_t)row * N + col] = acc[j][e];
        }
    }
}

// ---------------------------------------------------------------------------
// Sliding-window flash attention, one block = one head x 64 queries.
// 4 waves, each owns 16 query rows. KV processed in tiles of 32.
// Q tile lives in LDS (keeps the kernel under 256 VGPRs -> no MSB switching).
// K tile staged by TDM; V staged transposed by all threads (overlaps TDM).
// Per KV step per wave: 16 WMMA (QK^T) + 16 WMMA (P*V).
// ---------------------------------------------------------------------------
__global__ __launch_bounds__(128) void flash_attn_kernel(
    const __bf16* __restrict__ Q, const __bf16* __restrict__ K,
    const __bf16* __restrict__ V, __bf16* __restrict__ O) {
    constexpr float SCALE   = 0.0625f;       // 256^-0.5
    constexpr float CAP     = 50.0f;
    constexpr float INV_CAP = 1.0f / 50.0f;

    __shared__ __bf16 Qlds[64 * 256];        // [q][d]
    __shared__ __bf16 Klds[32 * 256];        // [kv][d] (TDM layout)
    __shared__ __bf16 Vlds[256 * 32];        // transposed: [d][kv]
    __shared__ __bf16 Plds[4][16 * 32];      // per-wave P tile [q][kv]

    const int wave = threadIdx.x >> 5;
    const int lane = threadIdx.x & 31;
    const int lh   = lane & 15;
    const int hf   = lane >> 4;
    const int h    = blockIdx.y;             // query head
    const int kvh  = h >> 1;                 // GQA: GROUPS = 2
    const int qb_block = blockIdx.x * 64;
    const int qb   = qb_block + wave * 16;
    const unsigned klds_off = (unsigned)(size_t)(void*)Klds;

    // ---- stage the 64x256 Q tile once ----
#pragma unroll
    for (int it = 0; it < 16; ++it) {
        int idx  = threadIdx.x + it * 128;   // 2048 chunks of 8 bf16
        int qrow = idx >> 5;
        int dch  = (idx & 31) * 8;
        *(uint4*)(Qlds + qrow * 256 + dch) =
            *(const uint4*)(Q + (size_t)(qb_block + qrow) * (HQ * HD) + h * HD + dch);
    }

    v8f acc[16];
#pragma unroll
    for (int n = 0; n < 16; ++n)
        acc[n] = (v8f){0, 0, 0, 0, 0, 0, 0, 0};
    float mrow[8], lrow[8];
#pragma unroll
    for (int r = 0; r < 8; ++r) { mrow[r] = -1e30f; lrow[r] = 0.0f; }

    int kv_lo = qb_block - (WIN - 1);
    if (kv_lo < 0) kv_lo = 0;
    kv_lo &= ~31;
    const int kv_hi = qb_block + 63;

    for (int kb = kv_lo; kb <= kv_hi; kb += 32) {
        __syncthreads();   // previous K/V tiles consumed (also fences Q tile)
#if HAVE_TDM
        if (wave == 0) {   // async DMA: K tile 32x256, row stride 2048 elems
            tdm_load_2d_bf16(klds_off,
                             K + (size_t)kb * (HKV * HD) + kvh * HD,
                             /*tile*/ HD, 32, /*tensor*/ HD, 32,
                             (unsigned long long)(HKV * HD));
        }
#else
#pragma unroll
        for (int it = 0; it < 8; ++it) {
            int idx = threadIdx.x + it * 128;        // 1024 chunks of 8
            int kvl = idx >> 5;
            int dch = (idx & 31) * 8;
            *(uint4*)(Klds + kvl * 256 + dch) =
                *(const uint4*)(K + (size_t)(kb + kvl) * (HKV * HD) + kvh * HD + dch);
        }
#endif
        // ---- stage V tile transposed: Vlds[d][kv] (overlaps the TDM) ----
#pragma unroll
        for (int it = 0; it < 16; ++it) {
            int idx = threadIdx.x + it * 128;        // 2048 chunks of 4
            int kvl = idx >> 6;
            int d0  = (idx & 63) * 4;
            const __bf16* vsrc = V + (size_t)(kb + kvl) * (HKV * HD) + kvh * HD + d0;
            bf2 a0 = *(const bf2*)(vsrc);
            bf2 a1 = *(const bf2*)(vsrc + 2);
            Vlds[(d0 + 0) * 32 + kvl] = a0.x;
            Vlds[(d0 + 1) * 32 + kvl] = a0.y;
            Vlds[(d0 + 2) * 32 + kvl] = a1.x;
            Vlds[(d0 + 3) * 32 + kvl] = a1.y;
        }
#if HAVE_TDM
        if (wave == 0) tdm_wait();
#endif
        __syncthreads();

        // ---- scores: S0 (kv cols 0..15), S1 (16..31), K-dim = D = 256 ----
        v8f s0 = (v8f){0, 0, 0, 0, 0, 0, 0, 0};
        v8f s1 = (v8f){0, 0, 0, 0, 0, 0, 0, 0};
#pragma unroll
        for (int f = 0; f < 8; ++f) {
            v16bf aq, b0, b1;
#pragma unroll
            for (int vg = 0; vg < 8; ++vg) {
                bf2 qv = *(const bf2*)(Qlds + (wave * 16 + lh) * 256 + f * 32 + a_koff(vg, hf));
                aq[2 * vg] = qv.x; aq[2 * vg + 1] = qv.y;
                int koff = f * 32 + b_koff(vg, hf);
                bf2 p0 = *(const bf2*)(Klds + lh * 256 + koff);
                bf2 p1 = *(const bf2*)(Klds + (lh + 16) * 256 + koff);
                b0[2 * vg] = p0.x; b0[2 * vg + 1] = p0.y;
                b1[2 * vg] = p1.x; b1[2 * vg + 1] = p1.y;
            }
            s0 = wmma_bf16(aq, b0, s0);
            s1 = wmma_bf16(aq, b1, s1);
        }

        // ---- softcap + mask + online softmax (half-wave reductions) ----
        float pv0[8], pv1[8], corr_r[8];
#pragma unroll
        for (int r = 0; r < 8; ++r) {
            int row = qb + (hf ? 8 + r : r);
            int j0  = kb + lh;
            int j1  = kb + 16 + lh;
            float x0 = CAP * tanhf(s0[r] * SCALE * INV_CAP);
            float x1 = CAP * tanhf(s1[r] * SCALE * INV_CAP);
            bool ok0 = (j0 <= row) && (row - j0 < WIN);
            bool ok1 = (j1 <= row) && (row - j1 < WIN);
            x0 = ok0 ? x0 : -1e30f;
            x1 = ok1 ? x1 : -1e30f;
            float mx = fmaxf(x0, x1);
#pragma unroll
            for (int off = 8; off >= 1; off >>= 1)
                mx = fmaxf(mx, __shfl_xor(mx, off, 32));
            float mnew = fmaxf(mrow[r], mx);
            float p0 = __expf(x0 - mnew);
            float p1 = __expf(x1 - mnew);
            float rs = p0 + p1;
#pragma unroll
            for (int off = 8; off >= 1; off >>= 1)
                rs += __shfl_xor(rs, off, 32);
            float corr = __expf(mrow[r] - mnew);
            lrow[r] = lrow[r] * corr + rs;
            mrow[r] = mnew;
            corr_r[r] = corr;
            pv0[r] = p0;
            pv1[r] = p1;
        }
        // rescale the O accumulators
#pragma unroll
        for (int n = 0; n < 16; ++n)
#pragma unroll
            for (int e = 0; e < 8; ++e)
                acc[n][e] = acc[n][e] * corr_r[e];

        // ---- reshape P (C layout) -> A-fragment via per-wave LDS ----
        __bf16* pl = Plds[wave];
#pragma unroll
        for (int r = 0; r < 8; ++r) {
            int prow = (hf ? 8 + r : r);
            pl[prow * 32 + lh]      = (__bf16)pv0[r];
            pl[prow * 32 + 16 + lh] = (__bf16)pv1[r];
        }
        asm volatile("s_wait_dscnt 0" ::: "memory");  // wave-local LDS RAW
        v16bf pf;
#pragma unroll
        for (int vg = 0; vg < 8; ++vg) {
            bf2 pp = *(const bf2*)(pl + lh * 32 + a_koff(vg, hf));
            pf[2 * vg]     = pp.x;
            pf[2 * vg + 1] = pp.y;
        }
        // ---- O += P (16x32) * V (32 x 256), 16 WMMAs along D ----
#pragma unroll
        for (int n = 0; n < 16; ++n) {
            v16bf vf;
#pragma unroll
            for (int vg = 0; vg < 8; ++vg) {
                bf2 vv = *(const bf2*)(Vlds + (n * 16 + lh) * 32 + b_koff(vg, hf));
                vf[2 * vg]     = vv.x;
                vf[2 * vg + 1] = vv.y;
            }
            acc[n] = wmma_bf16(pf, vf, acc[n]);
        }
    }

    // ---- epilogue: normalize and store bf16 [S, H*D] ----
#pragma unroll
    for (int n = 0; n < 16; ++n)
#pragma unroll
        for (int e = 0; e < 8; ++e) {
            int row = qb + (hf ? 8 + e : e);
            float inv_l = 1.0f / lrow[e];
            O[(size_t)row * (HQ * HD) + h * HD + n * 16 + lh] =
                (__bf16)(acc[n][e] * inv_l);
        }
}

// ---------------------------------------------------------------------------
// Host-side orchestration
// ---------------------------------------------------------------------------
extern "C" void kernel_launch(void* const* d_in, const int* in_sizes, int n_in,
                              void* d_out, int out_size, void* d_ws, size_t ws_size,
                              hipStream_t stream) {
    (void)in_sizes; (void)n_in; (void)out_size; (void)ws_size;
    const float* hs = (const float*)d_in[0];   // [S, E]
    const float* Wq = (const float*)d_in[1];   // [E, HQ*HD]
    const float* Wk = (const float*)d_in[2];   // [E, HKV*HD]
    const float* Wv = (const float*)d_in[3];   // [E, HKV*HD]
    const float* Wo = (const float*)d_in[4];   // [HQ*HD, E]
    const int*  pos = (const int*)d_in[5];     // [1, S]
    float*      out = (float*)d_out;           // [S, E]

    const size_t nX  = (size_t)S_LEN * EMB;
    const size_t nWq = (size_t)EMB * HQ * HD;
    const size_t nWk = (size_t)EMB * HKV * HD;
    const size_t nWo = (size_t)HQ * HD * EMB;
    const size_t nQ  = (size_t)S_LEN * HQ * HD;
    const size_t nK  = (size_t)S_LEN * HKV * HD;

    char* w = (char*)d_ws;
    __bf16* Xbf = (__bf16*)w;              w += nX  * 2;
    __bf16* Wqb = (__bf16*)w;              w += nWq * 2;
    __bf16* Wkb = (__bf16*)w;              w += nWk * 2;
    __bf16* Wvb = (__bf16*)w;              w += nWk * 2;
    __bf16* Wob = (__bf16*)w;              w += nWo * 2;
    __bf16* Qb  = (__bf16*)w;              w += nQ  * 2;
    __bf16* Kb  = (__bf16*)w;              w += nK  * 2;
    __bf16* Vb  = (__bf16*)w;              w += nK  * 2;
    __bf16* AOb = (__bf16*)w;              w += nQ  * 2;   // ~208 MB total

    auto cvt = [&](const float* src, __bf16* dst, size_t n) {
        int blocks = (int)((n / 4 + 255) / 256);
        cvt_f32_bf16<<<blocks, 256, 0, stream>>>(src, dst, (int)n);
    };
    cvt(hs, Xbf, nX);
    cvt(Wq, Wqb, nWq);
    cvt(Wk, Wkb, nWk);
    cvt(Wv, Wvb, nWk);
    cvt(Wo, Wob, nWo);

    // Q/K/V projections (bf16 out)
    gemm_wmma_bf16<true><<<dim3((HQ * HD) / 64, S_LEN / 128), 256, 0, stream>>>(
        Xbf, Wqb, (void*)Qb, S_LEN, HQ * HD, EMB);
    gemm_wmma_bf16<true><<<dim3((HKV * HD) / 64, S_LEN / 128), 256, 0, stream>>>(
        Xbf, Wkb, (void*)Kb, S_LEN, HKV * HD, EMB);
    gemm_wmma_bf16<true><<<dim3((HKV * HD) / 64, S_LEN / 128), 256, 0, stream>>>(
        Xbf, Wvb, (void*)Vb, S_LEN, HKV * HD, EMB);

    // RoPE (in place)
    rope_kernel<<<(S_LEN * HQ * 128) / 256, 256, 0, stream>>>(Qb, pos, HQ);
    rope_kernel<<<(S_LEN * HKV * 128) / 256, 256, 0, stream>>>(Kb, pos, HKV);

    // Sliding-window flash attention: grid (q-tiles, heads)
    flash_attn_kernel<<<dim3(S_LEN / 64, HQ), 128, 0, stream>>>(Qb, Kb, Vb, AOb);

    // Output projection (fp32 out, straight to d_out)
    gemm_wmma_bf16<false><<<dim3(EMB / 64, S_LEN / 128), 256, 0, stream>>>(
        AOb, Wob, (void*)out, S_LEN, EMB, HQ * HD);
}
